// RNNModel_68985764708473
// MI455X (gfx1250) — matrix operand order repaired and checked by reading, not compiled, not run on hardware
//
#include <hip/hip_runtime.h>
#include <hip/hip_bf16.h>

typedef unsigned short u16;
typedef __attribute__((ext_vector_type(16))) __bf16       v16bf;
typedef __attribute__((ext_vector_type(8)))  float        v8f;
typedef __attribute__((ext_vector_type(4)))  unsigned int u32x4;

#define T_STEPS 64
#define BATCH   512
#define NTOKS   512
#define NINPS   600
#define NHIDS   600
#define NBLK    6
#define BSZ     100
#define TOPKN   4
#define ATTD    340
#define DKD     64
#define DCD     32
#define KVW     404   /* DK + ATT columns of fused [Wk|Wv] GEMM   */
#define QKV2W   164   /* DC + DC + BS cols of [Wq2|Wk2|Wv2] GEMM  */

/* K-padded leading dims for bf16 A operands (multiples of 32) */
#define EPAD    608   /* 600 -> 608  (emb, hist)  */
#define RPAD    352   /* 340 -> 352  (read)       */
#define HPAD    128   /* 100 -> 128  (h, hn)      */

__device__ __forceinline__ u16 f2bf(float f) {
  union { float f; unsigned u; } x; x.f = f;
  unsigned r = x.u + 0x7FFFu + ((x.u >> 16) & 1u);  // round-to-nearest-even
  return (u16)(r >> 16);
}

union ABfrag {
  u16   s[16];
  u32x4 q[2];
  v16bf b;
};

// ---------------------------------------------------------------------------
// BF16 WMMA GEMM:  C(MxN) = A(M x Kpad bf16, zero-padded tail) * Bsw + bias
//  - M multiple of 16 (guaranteed by caller): no row guards.
//  - A: row-major, lda = Kpad (mult 32): two b128 loads per lane per chunk.
//  - Bsw: pre-swizzled fragment-major:
//      Bsw[(((kc*tilesN)+tn)*32 + lane)*16 + e]  = B[kc*32+half*16+e][tn*16+r]
//    two b128 loads per lane per chunk.
//  - One wave -> one 16x16 C tile; v_wmma_f32_16x16x32_bf16 per chunk.
// ---------------------------------------------------------------------------
__global__ __launch_bounds__(256) void gemm_bf16_wmma(
    const u16* __restrict__ A, int lda, int Kpad,
    const u16* __restrict__ Bsw, int Npad,
    const float* __restrict__ bias,
    float* __restrict__ Cf, u16* __restrict__ Cb, int ldc, int M, int N)
{
  const int wave   = blockIdx.x * 8 + (threadIdx.x >> 5);
  const int lane   = threadIdx.x & 31;
  const int tilesN = Npad >> 4;
  const int tilesM = M >> 4;
  const int tm = wave / tilesN;
  const int tn = wave - tm * tilesN;
  if (tm >= tilesM) return;                 // wave-uniform; EXEC all-1 at WMMA

  const int half = lane >> 4;
  const int r    = lane & 15;

  const u16*  Arow    = A + (size_t)(tm * 16 + r) * lda + half * 16;
  const u16*  Bbase   = Bsw + ((size_t)tn * 32 + lane) * 16;
  const size_t bstep  = (size_t)tilesN * 512;   // u16 elems per k-chunk

  v8f acc = {0.f,0.f,0.f,0.f,0.f,0.f,0.f,0.f};
  const int nk = Kpad >> 5;
  for (int kc = 0; kc < nk; ++kc) {
    ABfrag a, b;
    a.q[0] = *(const u32x4*)(Arow + kc * 32);
    a.q[1] = *(const u32x4*)(Arow + kc * 32 + 8);
    const u16* bp = Bbase + (size_t)kc * bstep;
    b.q[0] = *(const u32x4*)(bp);
    b.q[1] = *(const u32x4*)(bp + 8);
    acc = __builtin_amdgcn_wmma_f32_16x16x32_bf16(
        false, a.b, false, b.b, (short)0, acc, false, false);
  }

  const int  col   = tn * 16 + r;
  const bool colok = col < N;
  const float bv   = (bias && colok) ? bias[col] : 0.f;
#pragma unroll
  for (int i = 0; i < 8; ++i) {
    int row = tm * 16 + half * 8 + i;       // C: VGPR i -> row i + half*8
    if (colok) {
      float v = acc[i] + bv;
      if (Cb) Cb[(size_t)row * ldc + col] = f2bf(v);
      else    Cf[(size_t)row * ldc + col] = v;
    }
  }
}

// ---------------------------------------------------------------------------
// Prep kernels
// ---------------------------------------------------------------------------
__global__ void zero_u16_kernel(u16* p, int n) {
  int i = blockIdx.x * blockDim.x + threadIdx.x;
  if (i < n) p[i] = 0;
}

/* zero only the padded tail columns [n0, npad) of a rows x ld bf16 buffer */
__global__ void zero_pad_cols_kernel(u16* p, int ld, int n0, int npad, int rows) {
  int w = npad - n0;
  int i = blockIdx.x * blockDim.x + threadIdx.x;
  if (i >= rows * w) return;
  int rr = i / w, c = i - rr * w;
  p[(size_t)rr * ld + n0 + c] = 0;
}

__global__ void pack_w_kernel(const float* __restrict__ src, int K, int N,
                              u16* __restrict__ dst, int ldb, int coloff) {
  int i = blockIdx.x * blockDim.x + threadIdx.x;
  if (i >= K * N) return;
  int k = i / N, n = i - k * N;
  dst[(size_t)k * ldb + coloff + n] = f2bf(src[i]);
}

/* row-major padded (Kpad x Npad) -> fragment-major swizzle for the GEMM */
__global__ void swizzle_b_kernel(const u16* __restrict__ src, int Kpad, int Npad,
                                 u16* __restrict__ dst) {
  int i = blockIdx.x * blockDim.x + threadIdx.x;
  if (i >= Kpad * Npad) return;
  int e    = i & 15;
  int lane = (i >> 4) & 31;
  int rest = i >> 9;                       // kc*tilesN + tn
  int tilesN = Npad >> 4;
  int kc = rest / tilesN, tn = rest - kc * tilesN;
  int half = lane >> 4, r = lane & 15;
  int k   = kc * 32 + half * 16 + e;
  int col = tn * 16 + r;
  dst[i] = src[(size_t)k * Npad + col];
}

__global__ void f32_to_bf16_kernel(const float* __restrict__ src,
                                   u16* __restrict__ dst, int n) {
  int i = blockIdx.x * blockDim.x + threadIdx.x;
  if (i < n) dst[i] = f2bf(src[i]);
}

__global__ void copy_f32_kernel(const float* __restrict__ src,
                                float* __restrict__ dst, int n) {
  int i = blockIdx.x * blockDim.x + threadIdx.x;
  if (i < n) dst[i] = src[i];
}

// h = h_prev * mask[t]; fp32 flat (gates/blend) + bf16 (B*NB) x HPAD (GEMM A)
__global__ void mask_h_kernel(const float* __restrict__ hprev,
                              const float* __restrict__ mask,
                              float* __restrict__ hmask,
                              u16* __restrict__ hbf) {
  int i = blockIdx.x * blockDim.x + threadIdx.x;
  if (i >= BATCH * NHIDS) return;
  int b = i / NHIDS, w = i - b * NHIDS;
  int blk = w / BSZ, c = w - blk * BSZ;
  float v = hprev[i] * mask[b];
  hmask[i] = v;
  hbf[(size_t)(b * NBLK + blk) * HPAD + c] = f2bf(v);
}

// ---------------------------------------------------------------------------
// Step E1: softmax over {0,s} == sigmoid(s); top-4-of-6; read = p1 * vv (bf16)
// ---------------------------------------------------------------------------
__global__ __launch_bounds__(256) void attn_read_kernel(
    const float* __restrict__ q,        // (B*NB) x DK
    const float* __restrict__ kv,       // B x KVW  (kk | vv)
    u16*  __restrict__ read_bf,         // (B*NB) x RPAD
    float* __restrict__ bmask_ws,       // B x NB
    float* __restrict__ bmask_out)      // B x NB (this step's d_out slice)
{
  int b = blockIdx.x, tid = threadIdx.x;
  __shared__ float skk[DKD];
  __shared__ float scores[NBLK];
  __shared__ float p1s[NBLK];
  if (tid < DKD) skk[tid] = kv[(size_t)b * KVW + tid];
  __syncthreads();

  int w = tid >> 5, l = tid & 31;
  if (w < NBLK) {
    const float* qr = q + ((size_t)b * NBLK + w) * DKD;
    float p = qr[l] * skk[l] + qr[l + 32] * skk[l + 32];
    for (int off = 16; off; off >>= 1) p += __shfl_xor(p, off, 32);
    if (l == 0) scores[w] = p;
  }
  __syncthreads();

  if (tid == 0) {
    const float sc1 = 0.125f;            // 1/sqrt(64)
    bool used[NBLK]; float bm[NBLK];
    for (int n = 0; n < NBLK; ++n) {
      p1s[n] = 1.f / (1.f + __expf(-scores[n] * sc1));
      used[n] = false; bm[n] = 0.f;
    }
    for (int t = 0; t < TOPKN; ++t) {    // ties -> lowest index, like top_k
      int best = 0; float bv = -1e30f;
      for (int n = 0; n < NBLK; ++n)
        if (!used[n] && p1s[n] > bv) { bv = p1s[n]; best = n; }
      used[best] = true; bm[best] = 1.f;
    }
    for (int n = 0; n < NBLK; ++n) {
      bmask_ws[b * NBLK + n]  = bm[n];
      bmask_out[b * NBLK + n] = bm[n];
    }
  }
  __syncthreads();

  for (int i = tid; i < NBLK * ATTD; i += 256) {
    int n = i / ATTD, c = i - n * ATTD;
    read_bf[(size_t)(b * NBLK + n) * RPAD + c] =
        f2bf(p1s[n] * kv[(size_t)b * KVW + DKD + c]);
  }
}

// ---------------------------------------------------------------------------
// Step E2: GRU gates. xg already contains gru_b (GEMM bias).
// ---------------------------------------------------------------------------
__global__ void gru_kernel(const float* __restrict__ xg,
                           const float* __restrict__ hg,
                           const float* __restrict__ hmask,
                           float* __restrict__ hn_f,
                           u16*  __restrict__ hn_bf) {
  int i = blockIdx.x * blockDim.x + threadIdx.x;
  if (i >= BATCH * NBLK * BSZ) return;
  int row = i / BSZ, c = i - row * BSZ;
  int b = row / NBLK, blk = row - b * NBLK;
  const float* x = xg + (size_t)row * (3 * BSZ);
  const float* h = hg + (size_t)row * (3 * BSZ);
  float hb = hmask[(size_t)b * NHIDS + blk * BSZ + c];
  float r_ = 1.f / (1.f + __expf(-(x[c]       + h[c])));
  float z_ = 1.f / (1.f + __expf(-(x[BSZ + c] + h[BSZ + c])));
  float n_ = tanhf(x[2 * BSZ + c] + r_ * h[2 * BSZ + c]);
  float hn = (1.f - z_) * n_ + z_ * hb;
  hn_f[i] = hn;
  hn_bf[(size_t)row * HPAD + c] = f2bf(hn);
}

// ---------------------------------------------------------------------------
// Step E3: 6x6 block self-attention + top-k blend.
// Writes running fp32 h and the bf16 history row (A operand of decoder GEMM).
// ---------------------------------------------------------------------------
__global__ __launch_bounds__(256) void blockattn_kernel(
    const float* __restrict__ qkv2,     // (B*NB) x QKV2W : q2|k2|v2
    const float* __restrict__ hn_f,     // (B*NB) x BS
    const float* __restrict__ hmask,    // B x NHID
    const float* __restrict__ bmask_ws, // B x NB
    float* __restrict__ h_f32,          // B x NHID
    u16*  __restrict__ hist_bf_t)       // B x EPAD (this step's rows)
{
  int b = blockIdx.x, tid = threadIdx.x;
  __shared__ float sq[NBLK][DCD], sk[NBLK][DCD], p2[NBLK][NBLK];

  for (int i = tid; i < NBLK * 2 * DCD; i += 256) {
    int n = i / (2 * DCD), c = i - n * 2 * DCD;
    float v = qkv2[((size_t)b * NBLK + n) * QKV2W + c];
    if (c < DCD) sq[n][c] = v; else sk[n][c - DCD] = v;
  }
  __syncthreads();

  if (tid < NBLK * NBLK) {
    int n = tid / NBLK, m = tid - n * NBLK;
    float s = 0.f;
#pragma unroll
    for (int d = 0; d < DCD; ++d) s += sq[n][d] * sk[m][d];
    p2[n][m] = s * 0.17677669529663687f;   // 1/sqrt(32)
  }
  __syncthreads();

  if (tid < NBLK) {
    float mx = p2[tid][0];
    for (int m = 1; m < NBLK; ++m) mx = fmaxf(mx, p2[tid][m]);
    float e[NBLK], sum = 0.f;
    for (int m = 0; m < NBLK; ++m) { e[m] = __expf(p2[tid][m] - mx); sum += e[m]; }
    float inv = 1.f / sum;
    for (int m = 0; m < NBLK; ++m) p2[tid][m] = e[m] * inv;
  }
  __syncthreads();

  for (int i = tid; i < NHIDS; i += 256) {
    int n = i / BSZ, c = i - n * BSZ;
    float s = 0.f;
#pragma unroll
    for (int m = 0; m < NBLK; ++m)
      s += p2[n][m] * qkv2[((size_t)b * NBLK + m) * QKV2W + 2 * DCD + c];
    float hc = hn_f[((size_t)b * NBLK + n) * BSZ + c] + s;
    float mb = bmask_ws[b * NBLK + n];
    float hb = hmask[(size_t)b * NHIDS + i];
    float hnew = mb * hc + (1.f - mb) * hb;
    h_f32[(size_t)b * NHIDS + i]  = hnew;
    hist_bf_t[(size_t)b * EPAD + i] = f2bf(hnew);
  }
}

// ---------------------------------------------------------------------------
// Host side
// ---------------------------------------------------------------------------
static void launch_gemm(const u16* A, int lda, int Kpad,
                        const u16* Bsw, int Npad, const float* bias,
                        float* Cf, u16* Cb, int ldc, int M, int N,
                        hipStream_t s) {
  int tiles  = (M >> 4) * (Npad >> 4);
  int blocks = (tiles + 7) / 8;
  gemm_bf16_wmma<<<blocks, 256, 0, s>>>(A, lda, Kpad, Bsw, Npad, bias,
                                        Cf, Cb, ldc, M, N);
}

extern "C" void kernel_launch(void* const* d_in, const int* in_sizes, int n_in,
                              void* d_out, int out_size, void* d_ws, size_t ws_size,
                              hipStream_t stream) {
  (void)in_sizes; (void)n_in; (void)out_size; (void)ws_size;

  const float* in_input  = (const float*)d_in[0];
  const float* in_hidden = (const float*)d_in[1];
  const float* in_masks  = (const float*)d_in[2];
  const float* enc_W     = (const float*)d_in[3];
  const float* enc_b     = (const float*)d_in[4];
  const float* Wq        = (const float*)d_in[5];
  const float* Wk        = (const float*)d_in[6];
  const float* Wv        = (const float*)d_in[7];
  const float* Wx        = (const float*)d_in[8];
  const float* Wh        = (const float*)d_in[9];
  const float* gru_b     = (const float*)d_in[10];
  const float* Wq2       = (const float*)d_in[11];
  const float* Wk2       = (const float*)d_in[12];
  const float* Wv2       = (const float*)d_in[13];
  const float* dec_W     = (const float*)d_in[14];
  const float* dec_b     = (const float*)d_in[15];

  float* out_dec   = (float*)d_out;
  float* out_hT    = out_dec + (size_t)T_STEPS * BATCH * NTOKS;
  float* out_bmask = out_hT  + (size_t)BATCH * NHIDS;

  unsigned char* wsb = (unsigned char*)d_ws;
  size_t off = 0;
  auto alloc = [&](size_t bytes) -> void* {
    void* p = wsb + off;
    off = (off + bytes + 255) & ~(size_t)255;
    return p;
  };

  const int MROWS = T_STEPS * BATCH;          // 32768
  const int MBN   = BATCH * NBLK;             // 3072

  u16*   input_bf = (u16*)  alloc((size_t)MROWS * NTOKS * 2);  // lda 512
  u16*   emb_bf   = (u16*)  alloc((size_t)MROWS * EPAD * 2);   // lda 608
  u16*   hist_bf  = (u16*)  alloc((size_t)MROWS * EPAD * 2);   // lda 608
  float* h_f32    = (float*)alloc((size_t)BATCH * NHIDS * 4);
  float* hmask    = (float*)alloc((size_t)BATCH * NHIDS * 4);
  u16*   h_bf     = (u16*)  alloc((size_t)MBN * HPAD * 2);     // lda 128
  float* kv       = (float*)alloc((size_t)BATCH * KVW * 4);
  float* qbuf     = (float*)alloc((size_t)MBN * DKD * 4);
  u16*   read_bf  = (u16*)  alloc((size_t)MBN * RPAD * 2);     // lda 352
  float* xg       = (float*)alloc((size_t)MBN * 3 * BSZ * 4);
  float* hg       = (float*)alloc((size_t)MBN * 3 * BSZ * 4);
  float* hn_f     = (float*)alloc((size_t)MBN * BSZ * 4);
  u16*   hn_bf    = (u16*)  alloc((size_t)MBN * HPAD * 2);     // lda 128
  float* qkv2     = (float*)alloc((size_t)MBN * QKV2W * 4);
  float* bmask_ws = (float*)alloc((size_t)BATCH * NBLK * 4);

  // swizzled bf16 weights (Kpad x Npad each) + shared row-major staging
  u16* encW_sw  = (u16*)alloc((size_t)512 * 608 * 2);  // K512 N600
  u16* Wkv_sw   = (u16*)alloc((size_t)608 * 416 * 2);  // K600 N404
  u16* Wq_sw    = (u16*)alloc((size_t)128 *  64 * 2);  // K100 N64
  u16* Wx_sw    = (u16*)alloc((size_t)352 * 304 * 2);  // K340 N300
  u16* Wh_sw    = (u16*)alloc((size_t)128 * 304 * 2);  // K100 N300
  u16* Wqkv2_sw = (u16*)alloc((size_t)128 * 176 * 2);  // K100 N164
  u16* decW_sw  = (u16*)alloc((size_t)608 * 512 * 2);  // K600 N512
  u16* staging  = (u16*)alloc((size_t)608 * 512 * 2);  // largest Kpad*Npad

  auto zero = [&](u16* p, int n) {
    zero_u16_kernel<<<(n + 255) / 256, 256, 0, stream>>>(p, n);
  };
  auto pack = [&](const float* s, int K, int N, int ldb, int co) {
    pack_w_kernel<<<(K * N + 255) / 256, 256, 0, stream>>>(s, K, N, staging, ldb, co);
  };
  auto swz = [&](u16* dst, int Kpad, int Npad) {
    int n = Kpad * Npad;
    swizzle_b_kernel<<<(n + 255) / 256, 256, 0, stream>>>(staging, Kpad, Npad, dst);
  };
  auto padz = [&](u16* p, int ld, int n0, int rows) {
    int n = rows * (ld - n0);
    zero_pad_cols_kernel<<<(n + 255) / 256, 256, 0, stream>>>(p, ld, n0, ld, rows);
  };

  // ---- zero pad tails of all K-padded A operands (once per launch) ----
  padz(emb_bf,  EPAD, NINPS, MROWS);
  padz(hist_bf, EPAD, NHIDS, MROWS);
  padz(read_bf, RPAD, ATTD,  MBN);
  padz(h_bf,    HPAD, BSZ,   MBN);
  padz(hn_bf,   HPAD, BSZ,   MBN);

  // ---- pack + swizzle weights ----
  zero(staging, 512 * 608); pack(enc_W, 512, 600, 608, 0);
                            swz(encW_sw, 512, 608);
  zero(staging, 608 * 416); pack(Wk, 600,  64, 416, 0);
                            pack(Wv, 600, 340, 416, 64);
                            swz(Wkv_sw, 608, 416);
  zero(staging, 128 *  64); pack(Wq, 100, 64, 64, 0);
                            swz(Wq_sw, 128, 64);
  zero(staging, 352 * 304); pack(Wx, 340, 300, 304, 0);
                            swz(Wx_sw, 352, 304);
  zero(staging, 128 * 304); pack(Wh, 100, 300, 304, 0);
                            swz(Wh_sw, 128, 304);
  zero(staging, 128 * 176); pack(Wq2, 100,  32, 176, 0);
                            pack(Wk2, 100,  32, 176, 32);
                            pack(Wv2, 100, 100, 176, 64);
                            swz(Wqkv2_sw, 128, 176);
  zero(staging, 608 * 512); pack(dec_W, 600, 512, 512, 0);
                            swz(decW_sw, 608, 512);

  // ---- encoder: input -> bf16, then emb_bf = input @ enc_W + enc_b ----
  {
    int n = MROWS * NTOKS;
    f32_to_bf16_kernel<<<(n + 255) / 256, 256, 0, stream>>>(in_input, input_bf, n);
  }
  launch_gemm(input_bf, NTOKS, 512, encW_sw, 608, enc_b,
              nullptr, emb_bf, EPAD, MROWS, NINPS, stream);

  // ---- recurrent scan ----
  for (int t = 0; t < T_STEPS; ++t) {
    const float* hprev = (t == 0) ? in_hidden : h_f32;
    {
      int n = BATCH * NHIDS;
      mask_h_kernel<<<(n + 255) / 256, 256, 0, stream>>>(
          hprev, in_masks + (size_t)t * BATCH, hmask, h_bf);
    }
    // kk|vv = xt @ [Wk|Wv]
    launch_gemm(emb_bf + (size_t)t * BATCH * EPAD, EPAD, 608,
                Wkv_sw, 416, nullptr, kv, nullptr, KVW, BATCH, KVW, stream);
    // q = hb @ Wq
    launch_gemm(h_bf, HPAD, 128, Wq_sw, 64, nullptr,
                qbuf, nullptr, DKD, MBN, DKD, stream);
    attn_read_kernel<<<BATCH, 256, 0, stream>>>(
        qbuf, kv, read_bf, bmask_ws, out_bmask + (size_t)t * BATCH * NBLK);
    // xg = read @ Wx + gru_b
    launch_gemm(read_bf, RPAD, 352, Wx_sw, 304, gru_b,
                xg, nullptr, 3 * BSZ, MBN, 3 * BSZ, stream);
    // hg = hb @ Wh
    launch_gemm(h_bf, HPAD, 128, Wh_sw, 304, nullptr,
                hg, nullptr, 3 * BSZ, MBN, 3 * BSZ, stream);
    {
      int n = MBN * BSZ;
      gru_kernel<<<(n + 255) / 256, 256, 0, stream>>>(xg, hg, hmask, hn_f, hn_bf);
    }
    // q2|k2|v2 = hn @ [Wq2|Wk2|Wv2]
    launch_gemm(hn_bf, HPAD, 128, Wqkv2_sw, 176, nullptr,
                qkv2, nullptr, QKV2W, MBN, QKV2W, stream);
    blockattn_kernel<<<BATCH, 256, 0, stream>>>(
        qkv2, hn_f, hmask, bmask_ws, h_f32,
        hist_bf + (size_t)t * BATCH * EPAD);
  }

  // ---- decoder: dec = hist @ dec_W + dec_b ----
  launch_gemm(hist_bf, EPAD, 608, decW_sw, 512, dec_b,
              out_dec, nullptr, NTOKS, MROWS, NTOKS, stream);

  // ---- hT ----
  {
    int n = BATCH * NHIDS;
    copy_f32_kernel<<<(n + 255) / 256, 256, 0, stream>>>(h_f32, out_hT, n);
  }
}
